// Attention_69226282877525
// MI455X (gfx1250) — matrix operand order repaired and checked by reading, not compiled, not run on hardware
//
#include <hip/hip_runtime.h>
#include <hip/hip_bf16.h>

#define B_SZ 2
#define SEQ  2048
#define NH   16
#define HD   64
#define HID  1024
#define M_ROWS (B_SZ * SEQ)   // 4096

typedef __attribute__((ext_vector_type(16))) __bf16 bf16x16;
typedef __attribute__((ext_vector_type(8)))  __bf16 bf16x8;
typedef __attribute__((ext_vector_type(4)))  __bf16 bf16x4;
typedef __attribute__((ext_vector_type(8)))  float  v8f;
typedef __attribute__((ext_vector_type(4)))  int    i32x4;

// ---- gfx1250 async copy-to-LDS (ASYNCcnt path), guarded so absence of the
// ---- builtin degrades to the synchronous load+store staging.
#if defined(__gfx1250__) && __has_builtin(__builtin_amdgcn_global_load_async_to_lds_b128)
#define HAS_ASYNC_LDS 1
#else
#define HAS_ASYNC_LDS 0
#endif

__device__ __forceinline__ void wait_async0() {
#if defined(__gfx1250__)
#if __has_builtin(__builtin_amdgcn_s_wait_asynccnt)
    __builtin_amdgcn_s_wait_asynccnt(0);
#else
    asm volatile("s_wait_asynccnt 0" ::: "memory");
#endif
#endif
}

#if HAS_ASYNC_LDS
__device__ __forceinline__ void async_b128(const __bf16* g, __bf16* l) {
    // Builtin signature (from hipcc diagnostic): (int4 addrspace(1)*, int4 addrspace(3)*, int, int).
    // Go through integers: flat global addr == AS1 addr; LDS offset == low 32 bits of
    // the generic LDS address (ISA 10.2).
    __attribute__((address_space(1))) i32x4* gp =
        (__attribute__((address_space(1))) i32x4*)(unsigned long long)(uintptr_t)g;
    __attribute__((address_space(3))) i32x4* lp =
        (__attribute__((address_space(3))) i32x4*)(unsigned)(uintptr_t)l;
    __builtin_amdgcn_global_load_async_to_lds_b128(gp, lp, 0, 0);
}
#endif

__device__ __forceinline__ __bf16 f2bf(float f) {
    unsigned u = __builtin_bit_cast(unsigned, f);
    unsigned r = u + 0x7FFFu + ((u >> 16) & 1u);   // round-to-nearest-even
    unsigned short h = (unsigned short)(r >> 16);
    return __builtin_bit_cast(__bf16, h);
}

// Two 16-byte loads fused into one 16-element bf16 fragment register.
__device__ __forceinline__ bf16x16 load_frag16(const __bf16* p0, const __bf16* p1) {
    bf16x8 lo = *(const bf16x8*)p0;
    bf16x8 hi = *(const bf16x8*)p1;
    return __builtin_shufflevector(lo, hi, 0,1,2,3,4,5,6,7,8,9,10,11,12,13,14,15);
}

__device__ __forceinline__ v8f wmma_bf16(bf16x16 a, bf16x16 b, v8f c) {
    // (neg_a, A, neg_b, B, c_mod, C, reuse_a, reuse_b)
    return __builtin_amdgcn_wmma_f32_16x16x32_bf16(false, a, false, b, (short)0, c, false, false);
}

// ---------------- Phase 0a: fp32 -> bf16 conversion ----------------
__global__ void cvt_bf16_kernel(const float* __restrict__ src, __bf16* __restrict__ dst, int n4) {
    int i = blockIdx.x * blockDim.x + threadIdx.x;
    if (i >= n4) return;
    float4 v = ((const float4*)src)[i];
    bf16x4 o;
    o[0] = f2bf(v.x); o[1] = f2bf(v.y); o[2] = f2bf(v.z); o[3] = f2bf(v.w);
    *(bf16x4*)(dst + (size_t)i * 4) = o;
}

// Weight convert + transpose: dst[n][k] = src[k][n]  (coalesced writes)
__global__ void cvt_tr_kernel(const float* __restrict__ src, __bf16* __restrict__ dst) {
    int idx = blockIdx.x * blockDim.x + threadIdx.x;   // idx = n*HID + k
    int n = idx >> 10, k = idx & 1023;
    dst[idx] = f2bf(src[(size_t)k * HID + n]);
}

// ---------------- Phase 0b: transposed additive mask ----------------
// adT[b][t][f] = (1 - mask[b][f][t]) * -10000  (bf16; rows contiguous in f so the
// attention loop reads 8 row-adders with a single b128 instead of 8 b32 loads)
__global__ void mask_adder_kernel(const int* __restrict__ mask, __bf16* __restrict__ adT) {
    int idx = blockIdx.x * blockDim.x + threadIdx.x;   // b*SEQ*SEQ + t*SEQ + f
    int f = idx & 2047;
    int t = (idx >> 11) & 2047;
    int b = idx >> 22;
    int m = mask[((size_t)(b * SEQ + f)) * SEQ + t];
    adT[idx] = f2bf((1.0f - (float)m) * -10000.0f);
}

// ---------------- Phase 1: QKV projection GEMM ----------------
// Block: 8 waves sharing one 16-row A tile (staged once in LDS); each wave owns a
// 16x64 output tile. B fragments double-buffered in registers to hide load latency.
__global__ void qkv_gemm_kernel(const __bf16* __restrict__ X, const __bf16* __restrict__ Wt,
                                const float* __restrict__ bias, __bf16* __restrict__ out,
                                int transposedStore) {
    __shared__ __bf16 aS[16][HID];    // 32 KB: whole 16x1024 A tile
    int gwid = (blockIdx.x * blockDim.x + threadIdx.x) >> 5;
    int lane = threadIdx.x & 31;
    int half = lane >> 4, lr = lane & 15;

    int m0 = (gwid >> 4) * 16;        // same for all 8 waves of a block
    int n0 = (gwid & 15) * 64;

    // cooperative A-tile stage: 256 threads x 128 B
    {
        int row = threadIdx.x >> 4;
        int cb  = (threadIdx.x & 15) * 64;
        const __bf16* src = X + (size_t)(m0 + row) * HID + cb;
#if HAS_ASYNC_LDS
#pragma unroll
        for (int q = 0; q < 8; ++q)
            async_b128(src + q * 8, &aS[row][cb + q * 8]);
        wait_async0();
#else
#pragma unroll
        for (int q = 0; q < 8; ++q)
            *(bf16x8*)&aS[row][cb + q * 8] = *(const bf16x8*)(src + q * 8);
#endif
    }
    __syncthreads();

    auto loadB = [&](bf16x16* bfr, int k0) {
#pragma unroll
        for (int j = 0; j < 4; ++j) {
            const __bf16* bp = Wt + (size_t)(n0 + j * 16 + lr) * HID + k0 + half * 8;
            bfr[j] = load_frag16(bp, bp + 16);
        }
    };
    auto aFrag = [&](int k0) -> bf16x16 {
        const __bf16* p = &aS[lr][k0 + half * 8];
        return load_frag16(p, p + 16);
    };

    v8f acc[4] = {};
    bf16x16 bA[4], bB[4];
    loadB(bA, 0);
    for (int k0 = 0; k0 < HID; k0 += 64) {
        loadB(bB, k0 + 32);                       // prefetch while stage-0 WMMAs run
        bf16x16 a0 = aFrag(k0);
#pragma unroll
        for (int j = 0; j < 4; ++j) acc[j] = wmma_bf16(a0, bA[j], acc[j]);
        if (k0 + 64 < HID) loadB(bA, k0 + 64);    // prefetch for next iteration
        bf16x16 a1 = aFrag(k0 + 32);
#pragma unroll
        for (int j = 0; j < 4; ++j) acc[j] = wmma_bf16(a1, bB[j], acc[j]);
    }

#pragma unroll
    for (int j = 0; j < 4; ++j) {
        int n = n0 + j * 16 + lr;
        float bv = bias[n];
        int head = n >> 6, h = n & 63;
#pragma unroll
        for (int r = 0; r < 8; ++r) {
            int m = m0 + r + half * 8;
            int b = m >> 11, s = m & 2047;
            size_t addr = transposedStore
                ? ((size_t)((b * NH + head) * HD + h)) * SEQ + s
                : ((size_t)((b * NH + head) * SEQ + s)) * HD + h;
            out[addr] = f2bf(acc[j][r] + bv);
        }
    }
}

// ---------------- Phase 2: fused flash attention ----------------
// Block: 8 waves share one (b, head); K/V chunks staged in double-buffered LDS
// via async copy-to-LDS (ASYNCcnt) when available. Each wave owns 16 F-rows;
// online softmax over T in chunks of 32.
__global__ void attention_kernel(const __bf16* __restrict__ Q, const __bf16* __restrict__ K,
                                 const __bf16* __restrict__ Vt, const __bf16* __restrict__ adT,
                                 float* __restrict__ out) {
    __shared__ __bf16 kS[2][32][HD];    // [buf][t_local][h]    8 KB
    __shared__ __bf16 vS[2][HD][32];    // [buf][h][t_local]    8 KB
    __shared__ __bf16 plds[8][16][32];  // per-wave P staging   8 KB
    int gwid = (blockIdx.x * blockDim.x + threadIdx.x) >> 5;
    int w    = threadIdx.x >> 5;
    int lane = threadIdx.x & 31;
    int half = lane >> 4, lr = lane & 15;

    int f0   = (gwid & 127) * 16;
    int head = (gwid >> 7) & 15;
    int b    = gwid >> 11;

    const __bf16* Qh = Q   + ((size_t)(b * NH + head)) * SEQ * HD;
    const __bf16* Kh = K   + ((size_t)(b * NH + head)) * SEQ * HD;
    const __bf16* Vh = Vt  + ((size_t)(b * NH + head)) * HD * SEQ;
    const __bf16* aT = adT + (size_t)b * SEQ * SEQ;

    // Q tile 16x64 as two A-fragments (head-dim chunks 0..31, 32..63)
    const __bf16* qbase = Qh + (size_t)(f0 + lr) * HD + half * 8;
    bf16x16 qa0 = load_frag16(qbase,      qbase + 16);
    bf16x16 qa1 = load_frag16(qbase + 32, qbase + 48);

    // cooperative K/V chunk stage: 256 threads x (16 B K + 16 B V)
    auto fill = [&](int t0c, int buf) {
        int i  = threadIdx.x;
        int tl = i >> 3, ko = (i & 7) * 8;
        const __bf16* gk = Kh + (size_t)(t0c + tl) * HD + ko;
        __bf16*       lk = &kS[buf][tl][ko];
        int hl = i >> 2, to = (i & 3) * 8;
        const __bf16* gv = Vh + (size_t)hl * SEQ + t0c + to;
        __bf16*       lv = &vS[buf][hl][to];
#if HAS_ASYNC_LDS
        async_b128(gk, lk);
        async_b128(gv, lv);
#else
        *(bf16x8*)lk = *(const bf16x8*)gk;
        *(bf16x8*)lv = *(const bf16x8*)gv;
#endif
    };

    float mI[8], lI[8];
    v8f ctx[4] = {};
#pragma unroll
    for (int r = 0; r < 8; ++r) { mI[r] = -1e30f; lI[r] = 0.f; }

    fill(0, 0);
    for (int t0 = 0; t0 < SEQ; t0 += 32) {
        int buf = (t0 >> 5) & 1;
#if HAS_ASYNC_LDS
        wait_async0();                            // my async fills landed in LDS
#endif
        __syncthreads();                          // everyone's fill(buf) published
        if (t0 + 32 < SEQ) fill(t0 + 32, buf ^ 1);// stage next chunk during compute
#if !HAS_ASYNC_LDS
        if (t0 + 64 < SEQ) {
            __builtin_prefetch((const void*)(Kh + (size_t)(t0 + 64) * HD + (lane << 3)), 0, 0);
            __builtin_prefetch((const void*)(Vh + (size_t)(lane << 1) * SEQ + t0 + 64), 0, 0);
        }
#endif

        float sv[2][8];
#pragma unroll
        for (int c = 0; c < 2; ++c) {
            const __bf16* kb = &kS[buf][c * 16 + lr][half * 8];
            bf16x16 kb0 = load_frag16(kb,      kb + 16);   // h = 0..31
            bf16x16 kb1 = load_frag16(kb + 32, kb + 48);   // h = 32..63
            v8f s = {};
            s = wmma_bf16(qa0, kb0, s);
            s = wmma_bf16(qa1, kb1, s);
            int t = t0 + c * 16 + lr;
            bf16x8 ad = *(const bf16x8*)(aT + (size_t)t * SEQ + f0 + half * 8);
#pragma unroll
            for (int r = 0; r < 8; ++r)
                sv[c][r] = s[r] * 0.125f + (float)ad[r];   // scale 1/sqrt(64) + mask
        }
        // online softmax: per-row reductions across the 16-lane half-groups
#pragma unroll
        for (int r = 0; r < 8; ++r) {
            float x = fmaxf(sv[0][r], sv[1][r]);
            x = fmaxf(x, __shfl_xor(x, 1));
            x = fmaxf(x, __shfl_xor(x, 2));
            x = fmaxf(x, __shfl_xor(x, 4));
            x = fmaxf(x, __shfl_xor(x, 8));
            float mnew  = fmaxf(mI[r], x);
            float alpha = __expf(mI[r] - mnew);
            mI[r] = mnew;
            float p0 = __expf(sv[0][r] - mnew);
            float p1 = __expf(sv[1][r] - mnew);
            sv[0][r] = p0; sv[1][r] = p1;
            float rs = p0 + p1;
            rs += __shfl_xor(rs, 1);
            rs += __shfl_xor(rs, 2);
            rs += __shfl_xor(rs, 4);
            rs += __shfl_xor(rs, 8);
            lI[r] = lI[r] * alpha + rs;
#pragma unroll
            for (int j = 0; j < 4; ++j) ctx[j][r] *= alpha;
        }
        // D-layout P -> LDS -> A-fragment (wave-private, DS ops in-order)
#pragma unroll
        for (int c = 0; c < 2; ++c)
#pragma unroll
            for (int r = 0; r < 8; ++r)
                plds[w][r + half * 8][c * 16 + lr] = f2bf(sv[c][r]);
        const __bf16* pb = &plds[w][lr][half * 8];
        bf16x16 pfrag = load_frag16(pb, pb + 16);
        // context += P(16x32) x V(32x64) from LDS-staged V^T
#pragma unroll
        for (int j = 0; j < 4; ++j) {
            const __bf16* vb = &vS[buf][j * 16 + lr][half * 8];
            bf16x16 vfrag = load_frag16(vb, vb + 16);
            ctx[j] = wmma_bf16(pfrag, vfrag, ctx[j]);
        }
    }

#pragma unroll
    for (int j = 0; j < 4; ++j) {
        int h = j * 16 + lr;
#pragma unroll
        for (int r = 0; r < 8; ++r) {
            int f = f0 + r + half * 8;
            out[((size_t)(b * SEQ + f)) * HID + head * HD + h] = ctx[j][r] / lI[r];
        }
    }
}

// ---------------- host side ----------------
extern "C" void kernel_launch(void* const* d_in, const int* in_sizes, int n_in,
                              void* d_out, int out_size, void* d_ws, size_t ws_size,
                              hipStream_t stream) {
    const float* from = (const float*)d_in[0];
    const float* to   = (const float*)d_in[1];
    const int*   mask = (const int*)d_in[2];
    const float* Wq   = (const float*)d_in[3];
    const float* bq   = (const float*)d_in[4];
    const float* Wk   = (const float*)d_in[5];
    const float* bk   = (const float*)d_in[6];
    const float* Wv   = (const float*)d_in[7];
    const float* bv   = (const float*)d_in[8];
    float* out = (float*)d_out;

    // workspace carve (~62 MB, all regions fully rewritten every call)
    char* ws = (char*)d_ws;
    __bf16* Xf  = (__bf16*)(ws);                // 8 MB  from_tensor bf16
    __bf16* Xt  = (__bf16*)(ws +  8388608);     // 8 MB  to_tensor bf16
    __bf16* Wqt = (__bf16*)(ws + 16777216);     // 2 MB  Wq^T bf16
    __bf16* Wkt = (__bf16*)(ws + 18874368);     // 2 MB
    __bf16* Wvt = (__bf16*)(ws + 20971520);     // 2 MB
    __bf16* Qb  = (__bf16*)(ws + 23068672);     // 8 MB  [B][N][F][64]
    __bf16* Kb  = (__bf16*)(ws + 31457280);     // 8 MB  [B][N][T][64]
    __bf16* Vtb = (__bf16*)(ws + 39845888);     // 8 MB  [B][N][64][T]
    __bf16* adT = (__bf16*)(ws + 48234496);     // 16 MB [B][T][F] bf16 additive mask

    int n4 = M_ROWS * HID / 4;                  // 1048576 float4 groups
    cvt_bf16_kernel<<<n4 / 256, 256, 0, stream>>>(from, Xf, n4);
    cvt_bf16_kernel<<<n4 / 256, 256, 0, stream>>>(to,   Xt, n4);
    cvt_tr_kernel<<<4096, 256, 0, stream>>>(Wq, Wqt);
    cvt_tr_kernel<<<4096, 256, 0, stream>>>(Wk, Wkt);
    cvt_tr_kernel<<<4096, 256, 0, stream>>>(Wv, Wvt);
    mask_adder_kernel<<<32768, 256, 0, stream>>>(mask, adT);

    qkv_gemm_kernel<<<512, 256, 0, stream>>>(Xf, Wqt, bq, Qb,  0);
    qkv_gemm_kernel<<<512, 256, 0, stream>>>(Xt, Wkt, bk, Kb,  0);
    qkv_gemm_kernel<<<512, 256, 0, stream>>>(Xt, Wvt, bv, Vtb, 1);

    attention_kernel<<<512, 256, 0, stream>>>(Qb, Kb, Vtb, adT, out);
}
